// SimpleNet_37512244364141
// MI455X (gfx1250) — compile-verified, matrix operand order checked by reading, not run on hardware
//
#include <hip/hip_runtime.h>
#include <math.h>

// ---------------------------------------------------------------------------
// Problem constants (match reference)
// ---------------------------------------------------------------------------
#define NV 50000
#define NC 25000
#define NE 800000
#define NG 16
#define H  64

// ---------------------------------------------------------------------------
// CDNA5 WMMA plumbing (wave32, v_wmma_f32_16x16x32_bf16)
// ---------------------------------------------------------------------------
typedef __attribute__((ext_vector_type(16))) __bf16 v16bf;
typedef __attribute__((ext_vector_type(8)))  __bf16 v8bf;
typedef __attribute__((ext_vector_type(8)))  float  v8f;

#define WMMA_BF16(a, b, c) \
  __builtin_amdgcn_wmma_f32_16x16x32_bf16(false, (a), false, (b), (short)0, (c), false, false)

// Stage a row-major fp32 W[64,64] into LDS, pre-swizzled so that each lane's
// B-fragment (32x16 tile, K-half `half`, col-tile `t`) is one contiguous,
// 32-byte-aligned v16bf:  Wsw[(half*4 + t)*32 + lane].
// ISA B layout mirror of A: lane&15 = column, lane>>4 selects K-half-of-half,
// element e<8 -> k = k0+e, e>=8 -> k = k0+16+(e-8).
__device__ __forceinline__ void stage_w_swizzled(const float* __restrict__ W,
                                                 v16bf* Wsw, int tid, int nthr) {
  __bf16* wp = (__bf16*)Wsw;
  for (int d = tid; d < H * H; d += nthr) {
    int e = d & 15, lane = (d >> 4) & 31, t = (d >> 9) & 3, half = d >> 11;
    int col = (t << 4) + (lane & 15);
    int k   = (half << 5) + ((lane >> 4) << 3) + ((e & 8) << 1) + (e & 7);
    wp[d] = (__bf16)W[k * H + col];
  }
}

__device__ __forceinline__ v16bf bfrag_s(const v16bf* Wsw, int half, int t, int lane) {
  return Wsw[(half << 7) + (t << 5) + lane];   // single 32B LDS vector load
}

// A fragment (16x32 bf16) from a row-major fp32 [N,64] matrix (4x b128 loads).
__device__ __forceinline__ v16bf afrag_g(const float* __restrict__ X, int row0,
                                         int n, int kb, int lane) {
  int r = row0 + (lane & 15);
  if (r >= n) r = n - 1;                       // clamp; masked at store time
  const float* p = X + (size_t)r * H + kb + ((lane >> 4) << 3);
  float4 a0 = *(const float4*)(p);
  float4 a1 = *(const float4*)(p + 4);
  float4 b0 = *(const float4*)(p + 16);
  float4 b1 = *(const float4*)(p + 20);
  v16bf f;
  f[0] = (__bf16)a0.x; f[1] = (__bf16)a0.y; f[2]  = (__bf16)a0.z; f[3]  = (__bf16)a0.w;
  f[4] = (__bf16)a1.x; f[5] = (__bf16)a1.y; f[6]  = (__bf16)a1.z; f[7]  = (__bf16)a1.w;
  f[8] = (__bf16)b0.x; f[9] = (__bf16)b0.y; f[10] = (__bf16)b0.z; f[11] = (__bf16)b0.w;
  f[12] = (__bf16)b1.x; f[13] = (__bf16)b1.y; f[14] = (__bf16)b1.z; f[15] = (__bf16)b1.w;
  return f;
}

// A fragment from an LDS-resident bf16 [rows,64] tile (2x aligned 16B loads).
__device__ __forceinline__ v16bf afrag_l(const __bf16* T, int row0, int kb, int lane) {
  const __bf16* p = T + (row0 + (lane & 15)) * H + kb + ((lane >> 4) << 3);
  v8bf lo = *(const v8bf*)(p);
  v8bf hi = *(const v8bf*)(p + 16);
  return __builtin_shufflevector(lo, hi, 0, 1, 2, 3, 4, 5, 6, 7,
                                 8, 9, 10, 11, 12, 13, 14, 15);
}

// ---------------------------------------------------------------------------
// Generic fused GEMM: Y = act( X@W (+bias) (+X2@W2) (+S[row]*U[col]) (+Y) )
// Each wave: 16 rows x 64 cols.  Block: 256 thr = 8 waves = 128 rows.
// flags: bit0 = relu, bit1 = accumulate into existing Y
// ---------------------------------------------------------------------------
__global__ void __launch_bounds__(256) k_gemm64(
    const float* __restrict__ X,  const float* __restrict__ W,
    const float* __restrict__ B,
    const float* __restrict__ X2, const float* __restrict__ W2,
    const float* __restrict__ S,  const float* __restrict__ U,
    float* __restrict__ Y, int N, int flags) {
  __shared__ v16bf Wsw [256];
  __shared__ v16bf W2sw[256];
  int tid = threadIdx.x;
  stage_w_swizzled(W, Wsw, tid, 256);
  if (W2) stage_w_swizzled(W2, W2sw, tid, 256);
  __syncthreads();
  int wave = tid >> 5, lane = tid & 31;
  int row0 = blockIdx.x * 128 + wave * 16;
  if (row0 >= N) return;

  v8f acc[4] = {};
#pragma unroll
  for (int half = 0; half < 2; ++half) {
    v16bf a = afrag_g(X, row0, N, half << 5, lane);
#pragma unroll
    for (int t = 0; t < 4; ++t) acc[t] = WMMA_BF16(a, bfrag_s(Wsw, half, t, lane), acc[t]);
  }
  if (X2) {
#pragma unroll
    for (int half = 0; half < 2; ++half) {
      v16bf a = afrag_g(X2, row0, N, half << 5, lane);
#pragma unroll
      for (int t = 0; t < 4; ++t) acc[t] = WMMA_BF16(a, bfrag_s(W2sw, half, t, lane), acc[t]);
    }
  }
  // C/D layout: VGPR v, lanes 0-15 -> M=v ; lanes 16-31 -> M=v+8 ; N = lane&15 + 16*t
  int colb = lane & 15, rsh = (lane >> 4) << 3;
#pragma unroll
  for (int t = 0; t < 4; ++t) {
    int col = (t << 4) + colb;
    float bv = B ? B[col] : 0.f;
    float uv = U ? U[col] : 0.f;
#pragma unroll
    for (int v = 0; v < 8; ++v) {
      int row = row0 + rsh + v;
      if (row < N) {
        float val = acc[t][v] + bv;
        if (S) val += S[row] * uv;
        size_t o = (size_t)row * H + col;
        if (flags & 2) val += Y[o];
        if (flags & 1) val = fmaxf(val, 0.f);
        Y[o] = val;
      }
    }
  }
}

// ---------------------------------------------------------------------------
// Fused edge-encoder kernels.  Edge input is a scalar, so h(e) =
// W2 @ relu(e*w1+b1) + b2 is built per 64-edge tile entirely in LDS + WMMA,
// never materializing the [E,64] tensor (205 MB) in HBM.
// Block: 128 thr = 4 waves, 64 edges.
// ---------------------------------------------------------------------------
__global__ void __launch_bounds__(128) k_edge_stats(
    const float* __restrict__ ef, int E,
    const float* __restrict__ w1, const float* __restrict__ b1,
    const float* __restrict__ W2, const float* __restrict__ b2,
    float* __restrict__ stats /* [128] = per-channel sum, sumsq */) {
  __shared__ __attribute__((aligned(32))) __bf16 Tl[64 * H];
  __shared__ v16bf W2sw[256];
  __shared__ float bs[H], bq[H];
  int tid = threadIdx.x;
  int e0  = blockIdx.x * 64;
  __builtin_prefetch(ef + e0 + 64 + (tid & 15) * 4, 0, 3);   // next tile, streaming
  if (tid < H) { bs[tid] = 0.f; bq[tid] = 0.f; }
  stage_w_swizzled(W2, W2sw, tid, 128);
  for (int i = tid; i < 64 * H; i += 128) {
    int er = i >> 6, k = i & 63;
    int eg = e0 + er; if (eg >= E) eg = E - 1;
    Tl[i] = (__bf16)fmaxf(ef[eg] * w1[k] + b1[k], 0.f);
  }
  __syncthreads();
  int wave = tid >> 5, lane = tid & 31;
  int row0 = wave << 4;
  v8f acc[4] = {};
#pragma unroll
  for (int half = 0; half < 2; ++half) {
    v16bf a = afrag_l(Tl, row0, half << 5, lane);
#pragma unroll
    for (int t = 0; t < 4; ++t) acc[t] = WMMA_BF16(a, bfrag_s(W2sw, half, t, lane), acc[t]);
  }
  int colb = lane & 15, rsh = (lane >> 4) << 3;
#pragma unroll
  for (int t = 0; t < 4; ++t) {
    int col = (t << 4) + colb;
    float b2v = b2[col], s = 0.f, q = 0.f;
#pragma unroll
    for (int v = 0; v < 8; ++v) {
      int eg = e0 + row0 + rsh + v;
      if (eg < E) { float h = acc[t][v] + b2v; s += h; q += h * h; }
    }
    atomicAdd(&bs[col], s);
    atomicAdd(&bq[col], q);
  }
  __syncthreads();
  if (tid < H)          atomicAdd(&stats[tid], bs[tid]);
  else if (tid < 2 * H) atomicAdd(&stats[tid], bq[tid - H]);
}

__global__ void __launch_bounds__(128) k_edge_agg(
    const float* __restrict__ ef, const int* __restrict__ src,
    const int* __restrict__ dst, int E,
    const float* __restrict__ w1, const float* __restrict__ b1,
    const float* __restrict__ W2, const float* __restrict__ b2,
    const float* __restrict__ ss /* [128] = scale, shift */,
    const float* __restrict__ NSrc, float* __restrict__ AGG) {
  __shared__ __attribute__((aligned(32))) __bf16 Tl[64 * H];
  __shared__ v16bf W2sw[256];
  __shared__ int Sx[64], Dx[64];
  int tid = threadIdx.x;
  int e0  = blockIdx.x * 64;
  __builtin_prefetch(ef  + e0 + 64 + (tid & 15) * 4, 0, 3);
  __builtin_prefetch(src + e0 + 64 + (tid & 15) * 4, 0, 3);
  __builtin_prefetch(dst + e0 + 64 + (tid & 15) * 4, 0, 3);
  stage_w_swizzled(W2, W2sw, tid, 128);
  for (int i = tid; i < 64 * H; i += 128) {
    int er = i >> 6, k = i & 63;
    int eg = e0 + er; if (eg >= E) eg = E - 1;
    Tl[i] = (__bf16)fmaxf(ef[eg] * w1[k] + b1[k], 0.f);
  }
  if (tid < 64) {
    int eg = e0 + tid; if (eg >= E) eg = E - 1;
    Sx[tid] = src[eg]; Dx[tid] = dst[eg];
  }
  __syncthreads();
  int wave = tid >> 5, lane = tid & 31;
  int row0 = wave << 4;
  v8f acc[4] = {};
#pragma unroll
  for (int half = 0; half < 2; ++half) {
    v16bf a = afrag_l(Tl, row0, half << 5, lane);
#pragma unroll
    for (int t = 0; t < 4; ++t) acc[t] = WMMA_BF16(a, bfrag_s(W2sw, half, t, lane), acc[t]);
  }
  int colb = lane & 15, rsh = (lane >> 4) << 3;
#pragma unroll
  for (int t = 0; t < 4; ++t) {
    int col = (t << 4) + colb;
    float b2v = b2[col], sc = ss[col], sh = ss[H + col];
#pragma unroll
    for (int v = 0; v < 8; ++v) {
      int le = row0 + rsh + v;
      if (e0 + le < E) {
        float ee = sc * (acc[t][v] + b2v) + sh;                      // BN(edge enc)
        float m  = fmaxf(NSrc[(size_t)Sx[le] * H + col] + ee, 0.f);  // L2-resident gather
        atomicAdd(&AGG[(size_t)Dx[le] * H + col], m);                // L2-resident scatter
      }
    }
  }
}

// ---------------------------------------------------------------------------
// BatchNorm helpers (training-mode batch stats)
// ---------------------------------------------------------------------------
__global__ void __launch_bounds__(256) k_bn_stats(const float* __restrict__ X, int N,
                                                  float* __restrict__ stats) {
  __shared__ float bs[H], bq[H];
  int tid = threadIdx.x;
  if (tid < H) { bs[tid] = 0.f; bq[tid] = 0.f; }
  __syncthreads();
  int col   = tid & 63;
  int rbase = blockIdx.x * 256 + (tid >> 6);
  int rend  = min(N, (int)(blockIdx.x + 1) * 256);
  float s = 0.f, q = 0.f;
  for (int r = rbase; r < rend; r += 4) { float v = X[(size_t)r * H + col]; s += v; q += v * v; }
  atomicAdd(&bs[col], s);
  atomicAdd(&bq[col], q);
  __syncthreads();
  if (tid < H)          atomicAdd(&stats[tid], bs[tid]);
  else if (tid < 2 * H) atomicAdd(&stats[tid], bq[tid - H]);
}

__global__ void k_bn_finalize(const float* __restrict__ stats, float n,
                              const float* __restrict__ g, const float* __restrict__ beta,
                              float* __restrict__ ss) {
  int c = threadIdx.x;
  if (c < H) {
    float mu  = stats[c] / n;
    float var = stats[H + c] / n - mu * mu;
    float sc  = g[c] * rsqrtf(var + 1e-5f);
    ss[c]     = sc;
    ss[H + c] = beta[c] - mu * sc;
  }
}

__global__ void k_bn_apply(const float* __restrict__ X, const float* __restrict__ ss,
                           float* __restrict__ Y, int N) {
  int i = blockIdx.x * blockDim.x + threadIdx.x;
  if (i >= N * H) return;
  int c = i & 63;
  Y[i] = ss[c] * X[i] + ss[H + c];
}

// ---------------------------------------------------------------------------
// Small fused elementwise / reduction kernels
// ---------------------------------------------------------------------------
__global__ void k_lin_in2(const float* __restrict__ X /*[N,2]*/, const float* __restrict__ w,
                          const float* __restrict__ b, float* __restrict__ Y, int N) {
  int i = blockIdx.x * blockDim.x + threadIdx.x;
  if (i >= N * H) return;
  int r = i >> 6, c = i & 63;
  Y[i] = fmaxf(X[r * 2] * w[c] + X[r * 2 + 1] * w[H + c] + b[c], 0.f);
}

__global__ void k_lin_in1(const float* __restrict__ x, const float* __restrict__ sub,
                          const float* __restrict__ w, const float* __restrict__ b,
                          float* __restrict__ Y, int N) {
  int i = blockIdx.x * blockDim.x + threadIdx.x;
  if (i >= N * H) return;
  int r = i >> 6, c = i & 63;
  float t = x[r] - (sub ? sub[r] : 0.f);
  Y[i] = fmaxf(t * w[c] + b[c], 0.f);
}

__global__ void k_dot64_sigmoid(const float* __restrict__ X, const float* __restrict__ w,
                                const float* __restrict__ b, float* __restrict__ A, int N) {
  int r = blockIdx.x * blockDim.x + threadIdx.x;
  if (r >= N) return;
  float s = b[0];
#pragma unroll
  for (int k = 0; k < H; ++k) s += X[(size_t)r * H + k] * w[k];
  A[r] = 1.f / (1.f + expf(-s));
}

__global__ void k_scatter_a_e(const float* __restrict__ a, const float* __restrict__ ef,
                              const int* __restrict__ src, const int* __restrict__ dst,
                              float* __restrict__ out, int E) {
  int e = blockIdx.x * blockDim.x + threadIdx.x;
  if (e < E) atomicAdd(&out[dst[e]], a[src[e]] * ef[e]);
}

// order-preserving float<->uint map for atomic max
__device__ __forceinline__ unsigned fenc(float f) {
  unsigned u = __float_as_uint(f);
  return (u & 0x80000000u) ? ~u : (u | 0x80000000u);
}
__device__ __forceinline__ float fdec(unsigned e) {
  return __uint_as_float((e & 0x80000000u) ? (e & 0x7fffffffu) : ~e);
}

__global__ void k_seg_init(unsigned* __restrict__ menc, float* __restrict__ sums) {
  int i = blockIdx.x * blockDim.x + threadIdx.x;
  if (i < NG * H) { menc[i] = fenc(-INFINITY); sums[i] = 0.f; }
}
__global__ void k_segmax(const float* __restrict__ X, const int* __restrict__ idx,
                         unsigned* __restrict__ menc, int N) {
  int i = blockIdx.x * blockDim.x + threadIdx.x;
  if (i >= N * H) return;
  int r = i >> 6, c = i & 63;
  atomicMax(&menc[idx[r] * H + c], fenc(X[i]));
}
__global__ void k_segexp(float* __restrict__ X, const int* __restrict__ idx,
                         const unsigned* __restrict__ menc, float* __restrict__ sums, int N) {
  int i = blockIdx.x * blockDim.x + threadIdx.x;
  if (i >= N * H) return;
  int r = i >> 6, c = i & 63, g = idx[r];
  float e = expf(X[i] - fdec(menc[g * H + c]));
  X[i] = e;
  atomicAdd(&sums[g * H + c], e);
}
__global__ void k_segnorm(float* __restrict__ X, const int* __restrict__ idx,
                          const float* __restrict__ sums, int N) {
  int i = blockIdx.x * blockDim.x + threadIdx.x;
  if (i >= N * H) return;
  int r = i >> 6, c = i & 63;
  X[i] = X[i] / (sums[idx[r] * H + c] + 1e-16f);
}

// relu(l2norm(row)) — one wave per row, lane-permute reduction
__global__ void __launch_bounds__(256) k_rownorm_relu(const float* __restrict__ X,
                                                      float* __restrict__ Y, int N) {
  int gw   = (blockIdx.x * blockDim.x + threadIdx.x) >> 5;   // wave id == row
  int lane = threadIdx.x & 31;
  if (gw >= N) return;
  const float* p = X + (size_t)gw * H;
  float v0 = p[lane], v1 = p[lane + 32];
  float ss = v0 * v0 + v1 * v1;
#pragma unroll
  for (int o = 16; o > 0; o >>= 1) ss += __shfl_xor(ss, o, 32);
  float inv = 1.f / fmaxf(sqrtf(ss), 1e-12f);
  float* q = Y + (size_t)gw * H;
  q[lane]      = fmaxf(v0 * inv, 0.f);
  q[lane + 32] = fmaxf(v1 * inv, 0.f);
}

__global__ void k_lin4_logsoftmax(const float* __restrict__ X, const float* __restrict__ w,
                                  const float* __restrict__ b, float* __restrict__ out, int N) {
  int r = blockIdx.x * blockDim.x + threadIdx.x;
  if (r >= N) return;
  float z0 = b[0], z1 = b[1];
#pragma unroll
  for (int k = 0; k < H; ++k) {
    float x = X[(size_t)r * H + k];
    z0 += x * w[k * 2];
    z1 += x * w[k * 2 + 1];
  }
  float m = fmaxf(z0, z1);
  float l = m + logf(expf(z0 - m) + expf(z1 - m));
  out[r * 2]     = z0 - l;
  out[r * 2 + 1] = z1 - l;
}

// ---------------------------------------------------------------------------
// Host orchestration
// ---------------------------------------------------------------------------
namespace {
struct Lin { const float* b; const float* w; };
struct Enc { const float* beta; const float* g; Lin l1, l2; };
}  // namespace

extern "C" void kernel_launch(void* const* d_in, const int* in_sizes, int n_in,
                              void* d_out, int out_size, void* d_ws, size_t ws_size,
                              hipStream_t stream) {
  (void)in_sizes; (void)n_in; (void)out_size; (void)ws_size;

  const float* var_nf = (const float*)d_in[0];
  const float* con_nf = (const float*)d_in[1];
  const int*   ev_src = (const int*)d_in[2];
  const int*   ev_dst = (const int*)d_in[3];
  const int*   ec_src = (const int*)d_in[4];
  const int*   ec_dst = (const int*)d_in[5];
  const float* ef_var = (const float*)d_in[6];
  const float* ef_con = (const float*)d_in[7];
  const float* rhs    = (const float*)d_in[8];
  const int*   gidx   = (const int*)d_in[9];

  // params pytree: nested dicts in sorted-key order, each Linear = (b, w)
  int p = 10;
  auto L_  = [&]() { Lin l; l.b = (const float*)d_in[p++]; l.w = (const float*)d_in[p++]; return l; };
  auto Lnb = [&]() { Lin l; l.b = nullptr; l.w = (const float*)d_in[p++]; return l; };
  auto E_  = [&]() { Enc e; e.beta = (const float*)d_in[p++]; e.g = (const float*)d_in[p++];
                     e.l1 = L_(); e.l2 = L_(); return e; };
  Lin con_l1 = L_(), con_l2 = L_();                       // 'con_enc'
  struct Layer { Lin as1, as2; Enc cE; Lin cLl, cLr; Enc errE, jc, jv, vE; Lin vLl, vLr; } Ly[3];
  for (int i = 0; i < 3; ++i) {                           // 'layers'
    Ly[i].as1 = L_(); Ly[i].as2 = L_();                   //   assign
    Ly[i].cE  = E_();                                     //   c_edge_enc
    Ly[i].cLl = L_(); Ly[i].cLr = Lnb();                  //   c_lin_l / c_lin_r
    Ly[i].errE = E_();                                    //   err_enc
    Ly[i].jc  = E_();                                     //   joint_con
    Ly[i].jv  = E_();                                     //   joint_var
    Ly[i].vE  = E_();                                     //   v_edge_enc
    Ly[i].vLl = L_(); Ly[i].vLr = Lnb();                  //   v_lin_l / v_lin_r
  }
  Lin lin1 = L_(), lin2 = L_(), lin3 = L_(), lin4 = L_(); // 'lin1'..'lin4'
  Lin var_l1 = L_(), var_l2 = L_();                       // 'var_enc'

  // workspace carving (~135 MB fp32)
  float* base = (float*)d_ws;
  size_t off = 0;
  auto alloc = [&](size_t n) { float* r = base + off; off += n; return r; };
  float* xv[4]; for (int i = 0; i < 4; ++i) xv[i] = alloc((size_t)NV * H);
  float* xc[4]; for (int i = 0; i < 4; ++i) xc[i] = alloc((size_t)NC * H);
  float* bufA = alloc((size_t)NV * H);
  float* bufB = alloc((size_t)NV * H);
  float* bufC = alloc((size_t)NV * H);
  float* agg  = alloc((size_t)NV * H);
  float* errb = alloc((size_t)NC * H);
  float* av   = alloc(NV);
  float* tmp  = alloc(NC);
  float* stats = alloc(2 * H);
  float* ssA   = alloc(2 * H);
  float* ssB   = alloc(2 * H);
  unsigned* smax = (unsigned*)alloc(NG * H);
  float*    ssum = alloc(NG * H);

  auto gemm = [&](const float* X, const float* W, const float* B, const float* X2,
                  const float* W2, const float* S, const float* U, float* Y, int N, int fl) {
    k_gemm64<<<(N + 127) / 128, 256, 0, stream>>>(X, W, B, X2, W2, S, U, Y, N, fl);
  };
  auto zero = [&](void* ptr, size_t bytes) { hipMemsetAsync(ptr, 0, bytes, stream); };
  auto bn = [&](float* X, int N, const Enc& e, float* ss, float* Y) {
    zero(stats, 2 * H * sizeof(float));
    k_bn_stats<<<(N + 255) / 256, 256, 0, stream>>>(X, N, stats);
    k_bn_finalize<<<1, 64, 0, stream>>>(stats, (float)N, e.g, e.beta, ss);
    k_bn_apply<<<(N * H + 255) / 256, 256, 0, stream>>>(X, ss, Y, N);
  };
  auto edge_bn_stats = [&](const float* ef, const Enc& e, float* ss) {
    zero(stats, 2 * H * sizeof(float));
    k_edge_stats<<<(NE + 63) / 64, 128, 0, stream>>>(ef, NE, e.l1.w, e.l1.b, e.l2.w, e.l2.b, stats);
    k_bn_finalize<<<1, 64, 0, stream>>>(stats, (float)NE, e.g, e.beta, ss);
  };
  const int EW = (NV * H + 255) / 256;  // var-sized elementwise grid
  const int EC = (NC * H + 255) / 256;  // con-sized elementwise grid

  // ---- node encoders -------------------------------------------------------
  k_lin_in2<<<EW, 256, 0, stream>>>(var_nf, var_l1.w, var_l1.b, bufA, NV);
  gemm(bufA, var_l2.w, var_l2.b, nullptr, nullptr, nullptr, nullptr, xv[0], NV, 0);
  k_lin_in2<<<EC, 256, 0, stream>>>(con_nf, con_l1.w, con_l1.b, bufA, NC);
  gemm(bufA, con_l2.w, con_l2.b, nullptr, nullptr, nullptr, nullptr, xc[0], NC, 0);

  for (int l = 0; l < 3; ++l) {
    const Layer& Lp = Ly[l];
    const float* xvC = xv[l];
    const float* xcC = xc[l];

    // ---- ErrorLayer --------------------------------------------------------
    gemm(xvC, Lp.as1.w, Lp.as1.b, nullptr, nullptr, nullptr, nullptr, bufA, NV, 1);
    k_dot64_sigmoid<<<(NV + 255) / 256, 256, 0, stream>>>(bufA, Lp.as2.w, Lp.as2.b, av, NV);
    zero(tmp, NC * sizeof(float));
    k_scatter_a_e<<<(NE + 255) / 256, 256, 0, stream>>>(av, ef_var, ev_src, ev_dst, tmp, NE);
    k_lin_in1<<<EC, 256, 0, stream>>>(tmp, rhs, Lp.errE.l1.w, Lp.errE.l1.b, bufA, NC);
    gemm(bufA, Lp.errE.l2.w, Lp.errE.l2.b, nullptr, nullptr, nullptr, nullptr, bufB, NC, 0);
    bn(bufB, NC, Lp.errE, ssB, errb);
    k_seg_init<<<(NG * H + 255) / 256, 256, 0, stream>>>(smax, ssum);
    k_segmax<<<EC, 256, 0, stream>>>(errb, gidx, smax, NC);
    k_segexp<<<EC, 256, 0, stream>>>(errb, gidx, smax, ssum, NC);
    k_segnorm<<<EC, 256, 0, stream>>>(errb, gidx, ssum, NC);

    // ---- VarConBipartiteLayer (var -> con) ---------------------------------
    edge_bn_stats(ef_var, Lp.vE, ssA);
    // joint_var: [x_var | a] @ W[65,64]  ==  x_var@W[0:64] + a (x) W[64]
    gemm(xvC, Lp.jv.l1.w, Lp.jv.l1.b, nullptr, nullptr, av, Lp.jv.l1.w + 64 * H, bufA, NV, 1);
    gemm(bufA, Lp.jv.l2.w, Lp.jv.l2.b, nullptr, nullptr, nullptr, nullptr, bufB, NV, 0);
    bn(bufB, NV, Lp.jv, ssB, bufB);
    zero(agg, (size_t)NC * H * sizeof(float));
    k_edge_agg<<<(NE + 63) / 64, 128, 0, stream>>>(ef_var, ev_src, ev_dst, NE,
        Lp.vE.l1.w, Lp.vE.l1.b, Lp.vE.l2.w, Lp.vE.l2.b, ssA, bufB, agg);
    gemm(agg, Lp.vLl.w, Lp.vLl.b, xcC, Lp.vLr.w, nullptr, nullptr, bufC, NC, 0);
    k_rownorm_relu<<<(NC + 7) / 8, 256, 0, stream>>>(bufC, xc[l + 1], NC);

    // ---- ConVarBipartiteLayer (con -> var) ---------------------------------
    edge_bn_stats(ef_con, Lp.cE, ssA);
    // joint_con: [x_con_new | err] @ W[128,64]  == dual GEMM
    gemm(xc[l + 1], Lp.jc.l1.w, Lp.jc.l1.b, errb, Lp.jc.l1.w + 64 * H,
         nullptr, nullptr, bufA, NC, 1);
    gemm(bufA, Lp.jc.l2.w, Lp.jc.l2.b, nullptr, nullptr, nullptr, nullptr, bufB, NC, 0);
    bn(bufB, NC, Lp.jc, ssB, bufB);
    zero(agg, (size_t)NV * H * sizeof(float));
    k_edge_agg<<<(NE + 63) / 64, 128, 0, stream>>>(ef_con, ec_src, ec_dst, NE,
        Lp.cE.l1.w, Lp.cE.l1.b, Lp.cE.l2.w, Lp.cE.l2.b, ssA, bufB, agg);
    gemm(agg, Lp.cLl.w, Lp.cLl.b, xvC, Lp.cLr.w, nullptr, nullptr, bufC, NV, 0);
    k_rownorm_relu<<<(NV + 7) / 8, 256, 0, stream>>>(bufC, xv[l + 1], NV);
  }

  // ---- head: lin1(256->64) as 4 accumulated 64x64 GEMMs, lin2/3, lin4 ------
  gemm(xv[0], lin1.w,               lin1.b, nullptr, nullptr, nullptr, nullptr, bufA, NV, 0);
  gemm(xv[1], lin1.w + 1 * 64 * H, nullptr, nullptr, nullptr, nullptr, nullptr, bufA, NV, 2);
  gemm(xv[2], lin1.w + 2 * 64 * H, nullptr, nullptr, nullptr, nullptr, nullptr, bufA, NV, 2);
  gemm(xv[3], lin1.w + 3 * 64 * H, nullptr, nullptr, nullptr, nullptr, nullptr, bufA, NV, 3);
  gemm(bufA, lin2.w, lin2.b, nullptr, nullptr, nullptr, nullptr, bufB, NV, 1);
  gemm(bufB, lin3.w, lin3.b, nullptr, nullptr, nullptr, nullptr, bufA, NV, 1);
  k_lin4_logsoftmax<<<(NV + 255) / 256, 256, 0, stream>>>(bufA, lin4.w, lin4.b,
                                                          (float*)d_out, NV);
}
